// BestNetBilinear_17334488007397
// MI455X (gfx1250) — compile-verified, matrix operand order checked by reading, not run on hardware
//
#include <hip/hip_runtime.h>

// ---------------------------------------------------------------------------
// LRU residual block for MI455X (gfx1250), wave32 + v_wmma_f32_16x16x32_bf16.
// Weights staged in padded LDS; A-tiles register-resident where possible;
// depth-2/3 ping-pong tile buffers to keep loads ahead of the WMMA pipe.
// R = B*T = 131072 rows, all GEMMs are [R,256] x [256,256]^T.
// ---------------------------------------------------------------------------

typedef __attribute__((ext_vector_type(16))) __bf16 v16bf;
typedef __attribute__((ext_vector_type(8)))  __bf16 v8bf;
typedef __attribute__((ext_vector_type(8)))  float  v8f;

#define RR        131072      // 32 * 4096
#define DMODEL    256
#define NEG_SLOPE 0.01f
#define LN_EPS    1e-5f
#define LDS_LD    264         // padded LDS row stride (528B = 132 dwords -> +4 banks/row)

// ---- WMMA helper -----------------------------------------------------------
__device__ __forceinline__ v8f wmma_bf16(v16bf a, v16bf b, v8f c) {
    return __builtin_amdgcn_wmma_f32_16x16x32_bf16(false, a, false, b,
                                                   (short)0, c, false, false);
}

// 16-bit operand tile, 16 rows x 32 K, from global row-major (ld elements).
// ISA 7.12.2: lane L holds row (L&15), K = (L>>4)*8 + {0..7} and +16..23.
__device__ __forceinline__ v16bf load_tile(const __bf16* __restrict__ base,
                                           int ld, int r0, int k0, int lane) {
    const __bf16* p = base + (r0 + (lane & 15)) * ld + k0 + ((lane >> 4) << 3);
    v8bf lo = *(const v8bf*)p;
    v8bf hi = *(const v8bf*)(p + 16);
    v16bf t;
#pragma unroll
    for (int i = 0; i < 8; ++i) { t[i] = lo[i]; t[i + 8] = hi[i]; }
    return t;
}

// Same tile fetch but from the padded LDS copy (-> ds_load_b128).
__device__ __forceinline__ v16bf lds_tile(const __bf16* lds, int r0, int k0,
                                          int lane) {
    const __bf16* p = lds + (r0 + (lane & 15)) * LDS_LD + k0 + ((lane >> 4) << 3);
    v8bf lo = *(const v8bf*)p;
    v8bf hi = *(const v8bf*)(p + 16);
    v16bf t;
#pragma unroll
    for (int i = 0; i < 8; ++i) { t[i] = lo[i]; t[i + 8] = hi[i]; }
    return t;
}

// Cooperative copy of one 256x256 bf16 weight matrix into padded LDS.
__device__ __forceinline__ void stage_weight(__bf16* dst, const __bf16* __restrict__ src,
                                             int tid) {
    const __bf16* s = src + tid * DMODEL;
    __bf16* d = dst + tid * LDS_LD;
#pragma unroll
    for (int i = 0; i < 32; ++i)
        *(v8bf*)(d + i * 8) = *(const v8bf*)(s + i * 8);
}

__device__ __forceinline__ float lrelu(float v) {
    return v >= 0.0f ? v : NEG_SLOPE * v;
}

// ---- K0: weight fp32 -> bf16 ----------------------------------------------
__global__ __launch_bounds__(256) void k0_cvt(const float* __restrict__ s,
                                              __bf16* __restrict__ d, int n) {
    int i = blockIdx.x * 256 + threadIdx.x;
    if (i < n) d[i] = (__bf16)s[i];
}

// ---- K1: LN1 (affine) -> skip(f32);  leaky_relu -> u(bf16) -----------------
__global__ __launch_bounds__(256)
void k1_ln1(const float* __restrict__ x, const float* __restrict__ w,
            const float* __restrict__ b, float* __restrict__ skip,
            __bf16* __restrict__ u) {
    const int lane = threadIdx.x & 31;
    const int wave = threadIdx.x >> 5;
    const int row  = blockIdx.x * 8 + wave;
    const int base = row * DMODEL + lane * 8;

    float4 v0 = *(const float4*)(x + base);
    float4 v1 = *(const float4*)(x + base + 4);
    float vals[8] = {v0.x, v0.y, v0.z, v0.w, v1.x, v1.y, v1.z, v1.w};

    float s = 0.f, sq = 0.f;
#pragma unroll
    for (int i = 0; i < 8; ++i) { s += vals[i]; sq += vals[i] * vals[i]; }
#pragma unroll
    for (int m = 16; m >= 1; m >>= 1) {
        s  += __shfl_xor(s,  m, 32);
        sq += __shfl_xor(sq, m, 32);
    }
    const float mean = s * (1.0f / DMODEL);
    const float rstd = rsqrtf(sq * (1.0f / DMODEL) - mean * mean + LN_EPS);

    float yv[8];
    v8bf ub;
#pragma unroll
    for (int i = 0; i < 8; ++i) {
        const int c = lane * 8 + i;
        float t = (vals[i] - mean) * rstd * w[c] + b[c];
        yv[i] = t;
        ub[i] = (__bf16)lrelu(t);
    }
    *(float4*)(skip + base)     = make_float4(yv[0], yv[1], yv[2], yv[3]);
    *(float4*)(skip + base + 4) = make_float4(yv[4], yv[5], yv[6], yv[7]);
    *(v8bf*)(u + base) = ub;
}

// ---- K2: bu_{re,im} = gamma * (u @ B_{re,im}^T), bf16 out ------------------
// j-outer / k-inner with per-tile epilogue: accumulators are short-lived, so
// all 16 A-tiles (32 rows) stay register-resident; depth-2 LDS ping-pong.
__global__ __launch_bounds__(256)
void k2_bu(const __bf16* __restrict__ u, const __bf16* __restrict__ wre,
           const __bf16* __restrict__ wim, const float* __restrict__ gamma_log,
           __bf16* __restrict__ bure, __bf16* __restrict__ buim) {
    extern __shared__ __bf16 sm2[];
    __bf16* sre = sm2;
    __bf16* sim = sm2 + DMODEL * LDS_LD;
    const int tid = threadIdx.x;
    stage_weight(sre, wre, tid);
    stage_weight(sim, wim, tid);
    __syncthreads();

    const int lane = tid & 31;
    const int wave = tid >> 5;
    const int row0 = blockIdx.x * 128 + (wave >> 1) * 32;   // 32 rows / wave
    const int n0b  = (wave & 1) * 128;

    v16bf A0[8], A1[8];
#pragma unroll
    for (int k = 0; k < 8; ++k) {
        A0[k] = load_tile(u, DMODEL, row0,      k * 32, lane);
        A1[k] = load_tile(u, DMODEL, row0 + 16, k * 32, lane);
    }

    const int half = lane >> 4, nn = lane & 15;
#pragma unroll
    for (int j = 0; j < 8; ++j) {
        const int n0 = n0b + j * 16;
        v8f cr0 = (v8f)0.f, ci0 = (v8f)0.f, cr1 = (v8f)0.f, ci1 = (v8f)0.f;
        v16bf br[2], bi[2];
        br[0] = lds_tile(sre, n0, 0, lane);
        bi[0] = lds_tile(sim, n0, 0, lane);
#pragma unroll
        for (int k = 0; k < 8; ++k) {
            if (k < 7) {
                br[(k + 1) & 1] = lds_tile(sre, n0, (k + 1) * 32, lane);
                bi[(k + 1) & 1] = lds_tile(sim, n0, (k + 1) * 32, lane);
            }
            cr0 = wmma_bf16(A0[k], br[k & 1], cr0);
            ci0 = wmma_bf16(A0[k], bi[k & 1], ci0);
            cr1 = wmma_bf16(A1[k], br[k & 1], cr1);
            ci1 = wmma_bf16(A1[k], bi[k & 1], ci1);
        }
        const int col = n0 + nn;
        const float g = __expf(gamma_log[col]);
#pragma unroll
        for (int r = 0; r < 8; ++r) {
            const int rowA = row0 + r + 8 * half;
            const int rowB = rowA + 16;
            bure[rowA * DMODEL + col] = (__bf16)(cr0[r] * g);
            buim[rowA * DMODEL + col] = (__bf16)(ci0[r] * g);
            bure[rowB * DMODEL + col] = (__bf16)(cr1[r] * g);
            buim[rowB * DMODEL + col] = (__bf16)(ci1[r] * g);
        }
    }
}

// ---- K3: complex scan; stores hr and (-hi) as bf16 -------------------------
__global__ __launch_bounds__(256)
void k3_scan(const __bf16* __restrict__ bure, const __bf16* __restrict__ buim,
             const float* __restrict__ nu_log, const float* __restrict__ theta_log,
             __bf16* __restrict__ hro, __bf16* __restrict__ hino) {
    const int gid = blockIdx.x * 256 + threadIdx.x;  // 0..8191
    const int b = gid >> 8, n = gid & 255;
    const float lam = __expf(-__expf(nu_log[n]));
    const float th  = __expf(theta_log[n]);
    const float lr  = lam * __cosf(th);
    const float li  = lam * __sinf(th);

    float hr = 0.f, hi = 0.f;
    int idx = b * 4096 * DMODEL + n;
    for (int t = 0; t < 4096; ++t, idx += DMODEL) {
        __builtin_prefetch(&bure[idx + 8 * DMODEL], 0, 0);
        __builtin_prefetch(&buim[idx + 8 * DMODEL], 0, 0);
        const float br = (float)bure[idx];
        const float bi = (float)buim[idx];
        const float nr = __builtin_fmaf(lr, hr, __builtin_fmaf(-li, hi, br));
        const float ni = __builtin_fmaf(lr, hi, __builtin_fmaf( li, hr, bi));
        hr = nr; hi = ni;
        hro[idx]  = (__bf16)hr;
        hino[idx] = (__bf16)(-hi);
    }
}

// ---- K4: y = hr@Cre^T + (-hi)@Cim^T + u@Dm^T ; fused LN2(affine)+lrelu -----
// LDS: Cre + Cim staged (depth-2 ping-pong); Dm streamed global (depth-3).
__global__ __launch_bounds__(256)
void k4_y(const __bf16* __restrict__ hr, const __bf16* __restrict__ hin,
          const __bf16* __restrict__ u, const __bf16* __restrict__ wcre,
          const __bf16* __restrict__ wcim, const __bf16* __restrict__ wdm,
          const float* __restrict__ ln2w, const float* __restrict__ ln2b,
          __bf16* __restrict__ y) {
    extern __shared__ __bf16 sm4[];
    __bf16* scre = sm4;
    __bf16* scim = sm4 + DMODEL * LDS_LD;
    const int tid = threadIdx.x;
    stage_weight(scre, wcre, tid);
    stage_weight(scim, wcim, tid);
    __syncthreads();

    const int lane = tid & 31;
    const int wave = tid >> 5;
    const int row0 = blockIdx.x * 128 + wave * 16;

    v8f acc[16];
#pragma unroll
    for (int j = 0; j < 16; ++j) acc[j] = (v8f)0.f;

#pragma unroll
    for (int k0 = 0; k0 < DMODEL; k0 += 32) {
        v16bf a0 = load_tile(hr,  DMODEL, row0, k0, lane);
        v16bf a1 = load_tile(hin, DMODEL, row0, k0, lane);
        v16bf a2 = load_tile(u,   DMODEL, row0, k0, lane);
        v16bf bre[2], bim[2], bdm[3];
        bre[0] = lds_tile(scre, 0, k0, lane);
        bim[0] = lds_tile(scim, 0, k0, lane);
        bdm[0] = load_tile(wdm, DMODEL, 0,  k0, lane);
        bdm[1] = load_tile(wdm, DMODEL, 16, k0, lane);
#pragma unroll
        for (int j = 0; j < 16; ++j) {
            if (j < 15) {
                bre[(j + 1) & 1] = lds_tile(scre, (j + 1) * 16, k0, lane);
                bim[(j + 1) & 1] = lds_tile(scim, (j + 1) * 16, k0, lane);
            }
            if (j < 14)
                bdm[(j + 2) % 3] = load_tile(wdm, DMODEL, (j + 2) * 16, k0, lane);
            acc[j] = wmma_bf16(a0, bre[j & 1], acc[j]);
            acc[j] = wmma_bf16(a1, bim[j & 1], acc[j]);
            acc[j] = wmma_bf16(a2, bdm[j % 3], acc[j]);
        }
    }

    const int half = lane >> 4, nn = lane & 15;
#pragma unroll
    for (int r = 0; r < 8; ++r) {
        float s = 0.f, sq = 0.f;
#pragma unroll
        for (int j = 0; j < 16; ++j) { float v = acc[j][r]; s += v; sq += v * v; }
#pragma unroll
        for (int m = 8; m >= 1; m >>= 1) {
            s  += __shfl_xor(s,  m, 32);
            sq += __shfl_xor(sq, m, 32);
        }
        const float mean = s * (1.0f / DMODEL);
        const float rstd = rsqrtf(sq * (1.0f / DMODEL) - mean * mean + LN_EPS);
        const int row = row0 + r + 8 * half;
#pragma unroll
        for (int j = 0; j < 16; ++j) {
            const int col = j * 16 + nn;
            float v = (acc[j][r] - mean) * rstd * ln2w[col] + ln2b[col];
            y[row * DMODEL + col] = (__bf16)lrelu(v);
        }
    }
}

// ---- K5a: xl = LN_noaffine(y @ Wl^T + bl)  (bf16 out) ----------------------
// A-tiles register-resident; per-j k-inner loop, depth-2 LDS ping-pong.
__global__ __launch_bounds__(256)
void k5a_xl(const __bf16* __restrict__ y, const __bf16* __restrict__ wl,
            const float* __restrict__ bl, __bf16* __restrict__ xl) {
    extern __shared__ __bf16 sm5a[];
    const int tid = threadIdx.x;
    stage_weight(sm5a, wl, tid);
    __syncthreads();

    const int lane = tid & 31;
    const int wave = tid >> 5;
    const int row0 = blockIdx.x * 128 + wave * 16;

    v16bf A[8];
#pragma unroll
    for (int k = 0; k < 8; ++k) A[k] = load_tile(y, DMODEL, row0, k * 32, lane);

    v8f acc[16];
#pragma unroll
    for (int j = 0; j < 16; ++j) {
        v8f c = (v8f)0.f;
        v16bf b[2];
        b[0] = lds_tile(sm5a, j * 16, 0, lane);
#pragma unroll
        for (int k = 0; k < 8; ++k) {
            if (k < 7) b[(k + 1) & 1] = lds_tile(sm5a, j * 16, (k + 1) * 32, lane);
            c = wmma_bf16(A[k], b[k & 1], c);
        }
        acc[j] = c;
    }

    const int half = lane >> 4, nn = lane & 15;
    float bias[16];
#pragma unroll
    for (int j = 0; j < 16; ++j) bias[j] = bl[j * 16 + nn];

#pragma unroll
    for (int r = 0; r < 8; ++r) {
        float s = 0.f, sq = 0.f;
        float v[16];
#pragma unroll
        for (int j = 0; j < 16; ++j) {
            v[j] = acc[j][r] + bias[j]; s += v[j]; sq += v[j] * v[j];
        }
#pragma unroll
        for (int m = 8; m >= 1; m >>= 1) {
            s  += __shfl_xor(s,  m, 32);
            sq += __shfl_xor(sq, m, 32);
        }
        const float mean = s * (1.0f / DMODEL);
        const float rstd = rsqrtf(sq * (1.0f / DMODEL) - mean * mean + LN_EPS);
        const int row = row0 + r + 8 * half;
#pragma unroll
        for (int j = 0; j < 16; ++j)
            xl[row * DMODEL + j * 16 + nn] = (__bf16)((v[j] - mean) * rstd);
    }
}

// ---- K5b: xr GEMM + LN, gate with xl, LN4(affine), + skip -> out(f32) ------
__global__ __launch_bounds__(256)
void k5b_out(const __bf16* __restrict__ y, const __bf16* __restrict__ wr,
             const float* __restrict__ brv, const __bf16* __restrict__ xl,
             const float* __restrict__ skip, const float* __restrict__ ln4w,
             const float* __restrict__ ln4b, float* __restrict__ out) {
    extern __shared__ __bf16 sm5b[];
    const int tid = threadIdx.x;
    stage_weight(sm5b, wr, tid);
    __syncthreads();

    const int lane = tid & 31;
    const int wave = tid >> 5;
    const int row0 = blockIdx.x * 128 + wave * 16;

    v16bf A[8];
#pragma unroll
    for (int k = 0; k < 8; ++k) A[k] = load_tile(y, DMODEL, row0, k * 32, lane);

    v8f acc[16];
#pragma unroll
    for (int j = 0; j < 16; ++j) {
        v8f c = (v8f)0.f;
        v16bf b[2];
        b[0] = lds_tile(sm5b, j * 16, 0, lane);
#pragma unroll
        for (int k = 0; k < 8; ++k) {
            if (k < 7) b[(k + 1) & 1] = lds_tile(sm5b, j * 16, (k + 1) * 32, lane);
            c = wmma_bf16(A[k], b[k & 1], c);
        }
        acc[j] = c;
    }

    const int half = lane >> 4, nn = lane & 15;
    float bias[16];
#pragma unroll
    for (int j = 0; j < 16; ++j) bias[j] = brv[j * 16 + nn];

#pragma unroll
    for (int r = 0; r < 8; ++r) {
        const int row = row0 + r + 8 * half;
        float p[16];
        float s = 0.f, sq = 0.f;
#pragma unroll
        for (int j = 0; j < 16; ++j) {
            p[j] = acc[j][r] + bias[j]; s += p[j]; sq += p[j] * p[j];
        }
#pragma unroll
        for (int m = 8; m >= 1; m >>= 1) {
            s  += __shfl_xor(s,  m, 32);
            sq += __shfl_xor(sq, m, 32);
        }
        float mean = s * (1.0f / DMODEL);
        float rstd = rsqrtf(sq * (1.0f / DMODEL) - mean * mean + LN_EPS);

        float s2 = 0.f, sq2 = 0.f;
#pragma unroll
        for (int j = 0; j < 16; ++j) {
            float xrn = (p[j] - mean) * rstd;
            float xlv = (float)xl[row * DMODEL + j * 16 + nn];
            p[j] = xrn * xlv;
            s2 += p[j]; sq2 += p[j] * p[j];
        }
#pragma unroll
        for (int m = 8; m >= 1; m >>= 1) {
            s2  += __shfl_xor(s2,  m, 32);
            sq2 += __shfl_xor(sq2, m, 32);
        }
        const float mean2 = s2 * (1.0f / DMODEL);
        const float rstd2 = rsqrtf(sq2 * (1.0f / DMODEL) - mean2 * mean2 + LN_EPS);
#pragma unroll
        for (int j = 0; j < 16; ++j) {
            const int col = j * 16 + nn;
            out[row * DMODEL + col] =
                (p[j] - mean2) * rstd2 * ln4w[col] + ln4b[col] +
                skip[row * DMODEL + col];
        }
    }
}

// ---------------------------------------------------------------------------
extern "C" void kernel_launch(void* const* d_in, const int* in_sizes, int n_in,
                              void* d_out, int out_size, void* d_ws, size_t ws_size,
                              hipStream_t stream) {
    (void)in_sizes; (void)n_in; (void)out_size; (void)ws_size;

    const float* x         = (const float*)d_in[0];
    const float* ln1_w     = (const float*)d_in[1];
    const float* ln1_b     = (const float*)d_in[2];
    const float* nu_log    = (const float*)d_in[3];
    const float* theta_log = (const float*)d_in[4];
    const float* gamma_log = (const float*)d_in[5];
    const float* B_re      = (const float*)d_in[6];
    const float* B_im      = (const float*)d_in[7];
    const float* C_re      = (const float*)d_in[8];
    const float* C_im      = (const float*)d_in[9];
    const float* Dm        = (const float*)d_in[10];
    const float* ln2_w     = (const float*)d_in[11];
    const float* ln2_b     = (const float*)d_in[12];
    const float* Wl        = (const float*)d_in[13];
    const float* bl        = (const float*)d_in[14];
    const float* Wr        = (const float*)d_in[15];
    const float* br        = (const float*)d_in[16];
    const float* ln4_w     = (const float*)d_in[17];
    const float* ln4_b     = (const float*)d_in[18];
    float* out = (float*)d_out;

    // ---- workspace layout (bytes) ----
    char* ws = (char*)d_ws;
    const size_t F32_MAT = (size_t)RR * DMODEL * 4;   // 128 MiB
    const size_t BF_MAT  = (size_t)RR * DMODEL * 2;   //  64 MiB

    float*  skip  = (float*)ws;                               // [R,256] f32
    __bf16* u     = (__bf16*)(ws + F32_MAT);                  // [R,256]
    __bf16* bure  = (__bf16*)(ws + F32_MAT + 1 * BF_MAT);
    __bf16* buim  = (__bf16*)(ws + F32_MAT + 2 * BF_MAT);
    __bf16* hr    = (__bf16*)(ws + F32_MAT + 3 * BF_MAT);
    __bf16* hin   = (__bf16*)(ws + F32_MAT + 4 * BF_MAT);
    __bf16* yb    = bure;                                     // reuse (bu dead after scan)
    __bf16* xl    = buim;                                     // reuse
    __bf16* wbf   = (__bf16*)(ws + F32_MAT + 5 * BF_MAT);     // 7 x 128 KiB bf16 weights
    __bf16* wb_re = wbf + 0 * DMODEL * DMODEL;
    __bf16* wb_im = wbf + 1 * DMODEL * DMODEL;
    __bf16* wc_re = wbf + 2 * DMODEL * DMODEL;
    __bf16* wc_im = wbf + 3 * DMODEL * DMODEL;
    __bf16* wdm   = wbf + 4 * DMODEL * DMODEL;
    __bf16* wwl   = wbf + 5 * DMODEL * DMODEL;
    __bf16* wwr   = wbf + 6 * DMODEL * DMODEL;

    // dynamic LDS: padded weight copies (well under 320 KB/WGP)
    const size_t SM_ONE = (size_t)DMODEL * LDS_LD * 2;        // 132 KiB
    const size_t SM_TWO = 2 * SM_ONE;                         // 264 KiB

    // K0: convert the 7 weight matrices to bf16 (L2-resident thereafter)
    const int WN = DMODEL * DMODEL;
    const float* wsrc[7] = {B_re, B_im, C_re, C_im, Dm, Wl, Wr};
    __bf16*      wdst[7] = {wb_re, wb_im, wc_re, wc_im, wdm, wwl, wwr};
    for (int i = 0; i < 7; ++i)
        k0_cvt<<<WN / 256, 256, 0, stream>>>(wsrc[i], wdst[i], WN);

    // K1: LN1 + leaky_relu
    k1_ln1<<<RR / 8, 256, 0, stream>>>(x, ln1_w, ln1_b, skip, u);

    // K2: dual GEMM -> gamma-scaled bu_re / bu_im  (32 rows / wave)
    k2_bu<<<RR / 128, 256, SM_TWO, stream>>>(u, wb_re, wb_im, gamma_log, bure, buim);

    // K3: sequential complex scan (stores hr, -hi)
    k3_scan<<<32, 256, 0, stream>>>(bure, buim, nu_log, theta_log, hr, hin);

    // K4: triple GEMM + fused LN2 + leaky_relu -> y (bf16)
    k4_y<<<RR / 128, 256, SM_TWO, stream>>>(hr, hin, u, wc_re, wc_im, wdm,
                                            ln2_w, ln2_b, yb);

    // K5a: xl = LN(y @ Wl^T + bl)
    k5a_xl<<<RR / 128, 256, SM_ONE, stream>>>(yb, wwl, bl, xl);

    // K5b: xr GEMM + LN + gate + LN4 + residual -> out (f32)
    k5b_out<<<RR / 128, 256, SM_ONE, stream>>>(yb, wwr, br, xl, skip,
                                               ln4_w, ln4_b, out);
}
